// CalibrationNetwork_81329500717524
// MI455X (gfx1250) — matrix operand order repaired and checked by reading, not compiled, not run on hardware
//
#include <hip/hip_runtime.h>

// CDNA5 / gfx1250: wave32, WMMA 16x16x32 f16 -> f32.
typedef __attribute__((ext_vector_type(16))) _Float16 v16h;
typedef __attribute__((ext_vector_type(8)))  _Float16 v8h;
typedef __attribute__((ext_vector_type(8)))  float    v8f;

namespace {
constexpr int kB  = 16384;
constexpr int kJ  = 12;
constexpr int kQ  = 7;
constexpr int kO  = 5;      // options
constexpr int kIn = kO + 1; // 6 (bias + x)
constexpr int kH1 = 128;
constexpr int kH2 = 128;
constexpr int kMaxTiles = kB / 16 + kJ; // 1036 upper bound on same-judge 16-sample tiles

// workspace layout (int32 words)
constexpr int WS_COUNT = 0;   // [12] per-judge counts
constexpr int WS_SOFF  = 16;  // [13] per-judge sample offsets (prefix sum)
constexpr int WS_TOFF  = 32;  // [13] per-judge tile offsets
constexpr int WS_TOTAL = 48;  // [1]  total tiles
constexpr int WS_CUR   = 52;  // [12] scatter cursors
constexpr int WS_PERM  = 64;  // [16384] sample ids grouped by judge
}

// ---------------- binning kernels ----------------

__global__ void zero_k(int* ws) {
  int t = threadIdx.x;
  if (t < kJ) { ws[WS_COUNT + t] = 0; ws[WS_CUR + t] = 0; }
}

__global__ void count_k(const int* __restrict__ judge, int* __restrict__ ws) {
  int b = blockIdx.x * 256 + threadIdx.x;
  if (b < kB) {
    int j = judge[b];
    j = (j < 0) ? 0 : ((j >= kJ) ? kJ - 1 : j);
    atomicAdd(&ws[WS_COUNT + j], 1);
  }
}

__global__ void scan_k(int* ws) {
  if (threadIdx.x == 0) {
    int s = 0, t = 0;
    ws[WS_SOFF + 0] = 0;
    ws[WS_TOFF + 0] = 0;
    for (int j = 0; j < kJ; ++j) {
      s += ws[WS_COUNT + j];
      t += (ws[WS_COUNT + j] + 15) >> 4;
      ws[WS_SOFF + j + 1] = s;
      ws[WS_TOFF + j + 1] = t;
    }
    ws[WS_TOTAL] = t;
  }
}

__global__ void scatter_k(const int* __restrict__ judge, int* __restrict__ ws) {
  int b = blockIdx.x * 256 + threadIdx.x;
  if (b < kB) {
    int j = judge[b];
    j = (j < 0) ? 0 : ((j >= kJ) ? kJ - 1 : j);
    int p = atomicAdd(&ws[WS_CUR + j], 1);
    ws[WS_PERM + ws[WS_SOFF + j] + p] = b;
  }
}

// ---------------- WMMA fragment loaders (contiguous k order, same for A and B) ----------------
// A 16x32 f16: lane (m = lane&15, half = lane>>4); elem e: k = 16*half + e
__device__ __forceinline__ v16h a_frag(const _Float16* base, int stride, int m, int half, int kb) {
  const _Float16* row = base + m * stride + kb + half * 16;
  v8h lo = *(const v8h*)(row);        // k = kb + 16*half + 0..7
  v8h hi = *(const v8h*)(row + 8);    // k = kb + 16*half + 8..15
  return __builtin_shufflevector(lo, hi, 0,1,2,3,4,5,6,7,8,9,10,11,12,13,14,15);
}

// B 32x16 f16 from transposed [n][k] storage: lane (n = lane&15, half); elem e: k = kb + 16*half + e
__device__ __forceinline__ v16h b_frag(const _Float16* base, int stride, int n, int half, int kb) {
  const _Float16* p = base + n * stride + kb + half * 16;
  v8h lo = *(const v8h*)(p);
  v8h hi = *(const v8h*)(p + 8);
  return __builtin_shufflevector(lo, hi, 0,1,2,3,4,5,6,7,8,9,10,11,12,13,14,15);
}

// fast sigmoid: one v_exp + one v_rcp (avoid IEEE division expansion)
__device__ __forceinline__ float sigm(float v) {
  return __builtin_amdgcn_rcpf(1.0f + __expf(-v));
}

// ---------------- main compute kernel ----------------
// One workgroup = one tile of 16 samples sharing a judge; 8 wave32 per block.

__global__ __launch_bounds__(256) void calib_compute(
    const float* __restrict__ x,   const float* __restrict__ W1, const float* __restrict__ W1a,
    const float* __restrict__ W2,  const float* __restrict__ W2a,
    const float* __restrict__ V,   const float* __restrict__ Va,
    const int* __restrict__ ws,    float* __restrict__ out)
{
  // strides padded (40, 136 f16) -> 16B-aligned rows, no 16-way LDS bank conflicts
  __shared__ __align__(16) _Float16 s_w2t[kH1 * 136]; // [h][k], k = W2c row k+1 (bias peeled)
  __shared__ __align__(16) _Float16 s_w1t[kH1 * 40];  // [h][k], k<5 = W1c row k+1, else 0
  __shared__ __align__(16) _Float16 s_a1 [16 * 40];   // [s][k], k<5 = x, else 0
  __shared__ __align__(16) _Float16 s_z1 [16 * 136];  // [s][h] sigmoid(layer1)
  __shared__ __align__(16) _Float16 s_z2 [16 * 136];  // [s][h] sigmoid(layer2)
  __shared__ __align__(16) _Float16 s_vt [16 * 136];  // [n][k], n<5 = Vc col, else 0
  __shared__ float s_b1[kH1];
  __shared__ float s_b2[kH2];
  __shared__ float s_b3[16];
  __shared__ float s_lgp[4 * 16 * 16];   // per-wave partial logits (layer 3 K split)
  __shared__ int   s_idx[16];

  const int tid = threadIdx.x;
  const int g   = blockIdx.x;
  if (g >= ws[WS_TOTAL]) return;   // block-uniform

  // which judge / tile is this?
  const int* tOff = ws + WS_TOFF;
  int j = 0;
  #pragma unroll
  for (int t = 1; t < kJ; ++t) j += (g >= tOff[t]) ? 1 : 0;
  const int sStart = ws[WS_SOFF + j] + (g - tOff[j]) * 16;
  const int cnt    = min(16, ws[WS_SOFF + j + 1] - sStart);

  if (tid < 16) s_idx[tid] = ws[WS_PERM + sStart + min(tid, cnt - 1)];

  // combined W2 + W2_a[j], transposed to [h][k]; bias row (k==0) -> s_b2
  for (int e = tid; e < (kH1 + 1) * kH2; e += 256) {
    int k = e >> 7, h = e & 127;
    float v = W2[e] + W2a[j * (kH1 + 1) * kH2 + e];
    if (k == 0) s_b2[h] = v;
    else        s_w2t[h * 136 + (k - 1)] = (_Float16)v;
  }
  __syncthreads();

  const int lane = tid & 31, half = lane >> 4, ln16 = lane & 15;
  const int wv = tid >> 5;        // wave id 0..7
  const int h0 = wv * 16;         // this wave's N-tile

  for (int q = 0; q < kQ; ++q) {
    // ---- stage per-q operands into LDS ----
    for (int e = tid; e < 16 * 32; e += 256) {             // A1: x rows, zero-padded K
      int s = e >> 5, k = e & 31;
      float v = (k < kO) ? x[(s_idx[s] * kQ + q) * kO + k] : 0.0f;
      s_a1[s * 40 + k] = (_Float16)v;
    }
    for (int e = tid; e < kH1 * 32; e += 256) {            // W1c transposed [h][k]
      int h = e >> 5, k = e & 31;
      float v = 0.0f;
      if (k < kO) {
        int i = k + 1;
        v = W1[(q * kIn + i) * kH1 + h] + W1a[((j * kQ + q) * kIn + i) * kH1 + h];
      }
      s_w1t[h * 40 + k] = (_Float16)v;
    }
    if (tid < kH1)                                         // W1c bias row
      s_b1[tid] = W1[(q * kIn) * kH1 + tid] + W1a[((j * kQ + q) * kIn) * kH1 + tid];
    for (int e = tid; e < 16 * 128; e += 256) {            // Vc transposed [n][k]
      int n = e >> 7, k = e & 127;
      float v = 0.0f;
      if (n < kO)
        v = V[(q * (kH2 + 1) + (k + 1)) * kO + n] + Va[((j * kQ + q) * (kH2 + 1) + (k + 1)) * kO + n];
      s_vt[n * 136 + k] = (_Float16)v;
    }
    if (tid < 16) {                                        // Vc bias row (padded n)
      float v = (tid < kO)
                  ? (V[(q * (kH2 + 1)) * kO + tid] + Va[((j * kQ + q) * (kH2 + 1)) * kO + tid])
                  : 0.0f;
      s_b3[tid] = v;
    }
    __syncthreads();

    // ---- layer 1: z1 = sigmoid([1,x] @ W1c), K=5 padded to 32, bias via C-init ----
    {
      v16h a = a_frag(s_a1, 40, ln16, half, 0);
      v16h b = b_frag(s_w1t, 40, h0 + ln16, half, 0);
      v8f acc;
      float bias = s_b1[h0 + ln16];
      #pragma unroll
      for (int r = 0; r < 8; ++r) acc[r] = bias;
      acc = __builtin_amdgcn_wmma_f32_16x16x32_f16(false, a, false, b, (short)0, acc, false, false);
      #pragma unroll
      for (int r = 0; r < 8; ++r)
        s_z1[(half * 8 + r) * 136 + (h0 + ln16)] = (_Float16)sigm(acc[r]);
    }
    __syncthreads();

    // ---- layer 2: z2 = sigmoid([1,z1] @ W2c), K=128 in 4 chunks, bias via C-init ----
    {
      v8f acc;
      float bias = s_b2[h0 + ln16];
      #pragma unroll
      for (int r = 0; r < 8; ++r) acc[r] = bias;
      #pragma unroll
      for (int kb = 0; kb < kH1; kb += 32) {
        v16h a = a_frag(s_z1, 136, ln16, half, kb);
        v16h b = b_frag(s_w2t, 136, h0 + ln16, half, kb);
        acc = __builtin_amdgcn_wmma_f32_16x16x32_f16(false, a, false, b, (short)0, acc, false, false);
      }
      #pragma unroll
      for (int r = 0; r < 8; ++r)
        s_z2[(half * 8 + r) * 136 + (h0 + ln16)] = (_Float16)sigm(acc[r]);
    }
    __syncthreads();

    // ---- layer 3: logits = [1,z2] @ Vc, K split across waves 0..3 (one chunk each,
    //      independent accumulators -> no WMMA->WMMA RAW chain), partials to LDS ----
    if (wv < 4) {
      const int kb = wv * 32;
      v16h a = a_frag(s_z2, 136, ln16, half, kb);
      v16h b = b_frag(s_vt, 136, ln16, half, kb);
      v8f acc = {};
      acc = __builtin_amdgcn_wmma_f32_16x16x32_f16(false, a, false, b, (short)0, acc, false, false);
      #pragma unroll
      for (int r = 0; r < 8; ++r)
        s_lgp[(wv * 16 + half * 8 + r) * 16 + ln16] = acc[r];
    }
    __syncthreads();

    // ---- reduce partials + bias, softmax over 5 options, write f32 output ----
    if (tid < 16 && tid < cnt) {
      float l[kO];
      #pragma unroll
      for (int o = 0; o < kO; ++o) {
        l[o] = s_b3[o]
             + s_lgp[(0 * 16 + tid) * 16 + o]
             + s_lgp[(1 * 16 + tid) * 16 + o]
             + s_lgp[(2 * 16 + tid) * 16 + o]
             + s_lgp[(3 * 16 + tid) * 16 + o];
      }
      float m = l[0];
      #pragma unroll
      for (int o = 1; o < kO; ++o) m = fmaxf(m, l[o]);
      float e[kO], sum = 0.0f;
      #pragma unroll
      for (int o = 0; o < kO; ++o) { e[o] = __expf(l[o] - m); sum += e[o]; }
      float inv = __builtin_amdgcn_rcpf(sum);
      int base = (s_idx[tid] * kQ + q) * kO;
      #pragma unroll
      for (int o = 0; o < kO; ++o) out[base + o] = e[o] * inv;
    }
    __syncthreads();   // LDS reused next q
  }
}

// ---------------- launch ----------------

extern "C" void kernel_launch(void* const* d_in, const int* in_sizes, int n_in,
                              void* d_out, int out_size, void* d_ws, size_t ws_size,
                              hipStream_t stream) {
  const float* x   = (const float*)d_in[0];
  const int*   jid = (const int*)  d_in[1];   // judge_ids (int32 per harness convention)
  const float* W1  = (const float*)d_in[2];
  const float* W1a = (const float*)d_in[3];
  const float* W2  = (const float*)d_in[4];
  const float* W2a = (const float*)d_in[5];
  const float* V   = (const float*)d_in[6];
  const float* Va  = (const float*)d_in[7];
  float* out = (float*)d_out;
  int*   ws  = (int*)d_ws;

  zero_k   <<<1, 64, 0, stream>>>(ws);
  count_k  <<<kB / 256, 256, 0, stream>>>(jid, ws);
  scan_k   <<<1, 32, 0, stream>>>(ws);
  scatter_k<<<kB / 256, 256, 0, stream>>>(jid, ws);
  calib_compute<<<kMaxTiles, 256, 0, stream>>>(x, W1, W1a, W2, W2a, V, Va, ws, out);
}
